// DeepSeekMoE_90744069029991
// MI455X (gfx1250) — compile-verified, matrix operand order checked by reading, not run on hardware
//
#include <hip/hip_runtime.h>
#include <hip/hip_bf16.h>

// ---------------------------------------------------------------------------
// DeepSeek MoE (top-4 of 32 experts + shared expert) for gfx1250 / MI455X.
// Sparse expert dispatch + bf16 WMMA (v_wmma_f32_16x16x32_bf16).
// Fused gate/up/down SwiGLU per 32-token tile:
//   LDS: x-tile 32x1024 bf16 (64KB) + h-tile 32x2048 bf16 (128KB) = 192KB,
//   both XOR-swizzled at 32B-chunk granularity to kill bank conflicts.
// Shared expert and routed experts run in ONE launch (gridDim.y == kE+1).
// ---------------------------------------------------------------------------

typedef __attribute__((ext_vector_type(16))) __bf16 v16bf;
typedef __attribute__((ext_vector_type(8)))  float  v8f;

constexpr int kT = 1024;   // tokens (B*S)
constexpr int kD = 1024;   // model dim
constexpr int kF = 2048;   // ffn dim
constexpr int kE = 32;     // experts
constexpr int kK = 4;      // top-k
constexpr int kTM = 32;    // tokens per block tile

// workspace layout (bytes)
constexpr size_t OFF_ECNT = 0;                              // kE ints
constexpr size_t OFF_TOPI = 256;                            // kT*kK ints
constexpr size_t OFF_TOPW = OFF_TOPI + kT * kK * 4;         // kT*kK floats
constexpr size_t OFF_LIST = OFF_TOPW + kT * kK * 4;         // kE*kT ints
constexpr size_t OFF_WL   = OFF_LIST + (size_t)kE * kT * 4; // kE*kT floats
constexpr size_t OFF_XB   = OFF_WL + (size_t)kE * kT * 4;   // kT*kD bf16

// dynamic LDS partition (bytes)
constexpr size_t LDS_X = (size_t)kTM * kD * 2;     // 64 KB
constexpr size_t LDS_H = (size_t)kTM * kF * 2;     // 128 KB
constexpr size_t LDS_TOTAL = LDS_X + LDS_H;        // 192 KB (<320KB WGP LDS)

static __device__ inline v8f wmma_bf16(v16bf a, v16bf b, v8f c) {
  // D = A(16x32 bf16) * B(32x16 bf16) + C(16x16 f32)
  return __builtin_amdgcn_wmma_f32_16x16x32_bf16(
      /*neg_a=*/false, a, /*neg_b=*/false, b,
      /*c_mod=*/(short)0, c, /*reuse_a=*/false, /*reuse_b=*/false);
}

// Load 16 consecutive f32, convert to a bf16 fragment half-row (32B).
static __device__ inline v16bf load_frag_f32(const float* __restrict__ p) {
  v16bf r;
#pragma unroll
  for (int i = 0; i < 16; i += 4) {
    float4 v = *(const float4*)(p + i);
    r[i + 0] = (__bf16)v.x;
    r[i + 1] = (__bf16)v.y;
    r[i + 2] = (__bf16)v.z;
    r[i + 3] = (__bf16)v.w;
  }
  return r;
}

// ---------------------------------------------------------------------------
// x (f32) -> bf16, vectorized by 4
// ---------------------------------------------------------------------------
__global__ void k_cvt_bf16(const float* __restrict__ x, __bf16* __restrict__ xb, int n) {
  int i = (blockIdx.x * blockDim.x + threadIdx.x) * 4;
  if (i < n) {
    float4 v = *(const float4*)(x + i);
    xb[i + 0] = (__bf16)v.x;
    xb[i + 1] = (__bf16)v.y;
    xb[i + 2] = (__bf16)v.z;
    xb[i + 3] = (__bf16)v.w;
  }
}

// ---------------------------------------------------------------------------
// Router: one wave per token, lane e owns expert e.
// ---------------------------------------------------------------------------
__global__ void k_router(const float* __restrict__ x, const float* __restrict__ gw,
                         const float* __restrict__ bias,
                         int* __restrict__ topi, float* __restrict__ topw) {
  const int t = blockIdx.x;
  const int e = threadIdx.x;  // 0..31
  const float* xr = x + (size_t)t * kD;
  const float* wr = gw + (size_t)e * kD;
  float acc = 0.f;
  for (int k = 0; k < kD; k += 4) {
    float4 xv = *(const float4*)(xr + k);   // broadcast across lanes
    float4 wv = *(const float4*)(wr + k);
    acc = fmaf(xv.x, wv.x, fmaf(xv.y, wv.y, fmaf(xv.z, wv.z, fmaf(xv.w, wv.w, acc))));
  }
  float val = acc + bias[e];

  float tv[kK];
  int   ti[kK];
#pragma unroll
  for (int i = 0; i < kK; ++i) {
    float m = val;
#pragma unroll
    for (int off = 16; off > 0; off >>= 1) m = fmaxf(m, __shfl_xor(m, off, 32));
    unsigned msk = (unsigned)__ballot(val == m);
    int owner = __ffs(msk) - 1;   // lowest lane wins ties (lax.top_k order)
    tv[i] = m;
    ti[i] = owner;
    if (e == owner) val = -3.402823466e38f;
  }
  float s = 0.f, w[kK];
#pragma unroll
  for (int i = 0; i < kK; ++i) { w[i] = __expf(tv[i] - tv[0]); s += w[i]; }
  if (e < kK) {
    topi[t * kK + e] = ti[e];
    topw[t * kK + e] = w[e] / s;
  }
}

// ---------------------------------------------------------------------------
// Deterministic per-expert token lists + counts + load_balance.
// ---------------------------------------------------------------------------
__global__ void k_build(const int* __restrict__ topi, const float* __restrict__ topw,
                        int* __restrict__ ecnt, int* __restrict__ list,
                        float* __restrict__ wl, float* __restrict__ counts,
                        float* __restrict__ lb) {
  const int e = threadIdx.x;
  int c = 0;
  for (int t = 0; t < kT; ++t) {
#pragma unroll
    for (int j = 0; j < kK; ++j) {
      if (topi[t * kK + j] == e) {
        list[e * kT + c] = t;
        wl[e * kT + c]   = topw[t * kK + j];
        ++c;
      }
    }
  }
  ecnt[e] = c;
  float fc = (float)c;
  counts[e] = fc;
  float s = fc, s2 = fc * fc;
#pragma unroll
  for (int off = 16; off > 0; off >>= 1) {
    s  += __shfl_xor(s, off, 32);
    s2 += __shfl_xor(s2, off, 32);
  }
  if (e == 0) {
    float mean = s * (1.f / 32.f);
    float var = fmaxf(s2 * (1.f / 32.f) - mean * mean, 0.f);
    lb[0] = sqrtf(var) / (mean + 1e-6f);
  }
}

// ---------------------------------------------------------------------------
// Fused SwiGLU expert tile: 32 tokens x full F, 8 waves / 256 threads.
// blockIdx.y in [0,kE) -> routed expert; blockIdx.y == kE -> shared expert.
// ---------------------------------------------------------------------------
__global__ __launch_bounds__(256) void k_expert(
    const __bf16* __restrict__ xb,
    const float* __restrict__ Wg0, const float* __restrict__ Wu0,
    const float* __restrict__ Wd0,
    const float* __restrict__ Wsg, const float* __restrict__ Wsu,
    const float* __restrict__ Wsd,
    const int* __restrict__ list0, const float* __restrict__ wl0,
    const int* __restrict__ ecnt, float* __restrict__ out) {
  const int tile = blockIdx.x;
  const int ey = blockIdx.y;
  const bool is_shared = (ey == kE);

  const float* Wg;
  const float* Wu;
  const float* Wd;
  const int* list = nullptr;
  const float* wl = nullptr;
  int cnt;
  if (is_shared) {
    Wg = Wsg; Wu = Wsu; Wd = Wsd; cnt = kT;
  } else {
    cnt = ecnt[ey];
    if (tile * kTM >= cnt) return;  // uniform early exit
    const size_t wo = (size_t)ey * kF * kD;
    Wg = Wg0 + wo; Wu = Wu0 + wo; Wd = Wd0 + wo;
    list = list0 + (size_t)ey * kT;
    wl   = wl0 + (size_t)ey * kT;
  }

  extern __shared__ char smem[];
  __bf16* xs = (__bf16*)smem;             // [32][1024] swizzled
  __bf16* hs = (__bf16*)(smem + LDS_X);   // [32][2048] swizzled

  const int lane = threadIdx.x & 31;
  const int wave = threadIdx.x >> 5;
  const int lm = lane & 15;   // row (A/C) or column (B/C) index within a tile
  const int lh = lane >> 4;   // K-half selector

  // ---- stage gathered x tile into LDS (32 rows x 64 chunks of 32B) ----
  for (int i = threadIdx.x; i < kTM * (kD / 16); i += 256) {
    const int r = i >> 6;       // row 0..31
    const int c = i & 63;       // 16-elem chunk 0..63
    const int row = tile * kTM + r;
    const int tok = is_shared ? row : list[row < cnt ? row : cnt - 1];
    const v16bf v = *(const v16bf*)(xb + (size_t)tok * kD + c * 16);
    *(v16bf*)(&xs[r * kD + ((c ^ (r & 7)) << 4)]) = v;
  }

  // Per-lane token/scale for the C rows this lane owns (m = mt*16 + r + 8*lh)
  int   tokS[2][8];
  float sclS[2][8];
#pragma unroll
  for (int mt = 0; mt < 2; ++mt) {
#pragma unroll
    for (int r = 0; r < 8; ++r) {
      const int m = mt * 16 + r + lh * 8;
      const int row = tile * kTM + m;
      if (is_shared)      { tokS[mt][r] = row;       sclS[mt][r] = 1.f; }
      else if (row < cnt) { tokS[mt][r] = list[row]; sclS[mt][r] = wl[row]; }
      else                { tokS[mt][r] = -1;        sclS[mt][r] = 0.f; }
    }
  }

  __syncthreads();

  const v8f vzero = {0.f, 0.f, 0.f, 0.f, 0.f, 0.f, 0.f, 0.f};

  // ---------------- Phase A: h tile into LDS ----------------
  // wave owns columns [wave*256, wave*256+256): 8 groups of 2 n-tiles
#pragma unroll 1
  for (int g = 0; g < 8; ++g) {
    const int n0 = wave * 256 + g * 32;
    v8f accG[2][2], accU[2][2];   // [mt][j]
#pragma unroll
    for (int mt = 0; mt < 2; ++mt)
#pragma unroll
      for (int j = 0; j < 2; ++j) { accG[mt][j] = vzero; accU[mt][j] = vzero; }

#pragma unroll 2
    for (int k0 = 0; k0 < kD; k0 += 32) {
      const int ca = ((k0 >> 4) + lh) ^ (lm & 7);
      const v16bf a0 = *(const v16bf*)(&xs[lm * kD + (ca << 4)]);
      const v16bf a1 = *(const v16bf*)(&xs[(16 + lm) * kD + (ca << 4)]);
#pragma unroll
      for (int j = 0; j < 2; ++j) {
        const float* gp = Wg + (size_t)(n0 + j * 16 + lm) * kD + lh * 16 + k0;
        const float* up = Wu + (size_t)(n0 + j * 16 + lm) * kD + lh * 16 + k0;
        const v16bf bg = load_frag_f32(gp);
        const v16bf bu = load_frag_f32(up);
        accG[0][j] = wmma_bf16(a0, bg, accG[0][j]);
        accG[1][j] = wmma_bf16(a1, bg, accG[1][j]);
        accU[0][j] = wmma_bf16(a0, bu, accU[0][j]);
        accU[1][j] = wmma_bf16(a1, bu, accU[1][j]);
      }
    }

#pragma unroll
    for (int mt = 0; mt < 2; ++mt) {
#pragma unroll
      for (int j = 0; j < 2; ++j) {
        const int chunk = (n0 >> 4) + j;
#pragma unroll
        for (int r = 0; r < 8; ++r) {
          const int m = mt * 16 + r + lh * 8;
          const float gg = accG[mt][j][r];
          const float uu = accU[mt][j][r];
          const float h = gg / (1.f + __expf(-gg)) * uu;   // silu(g)*u
          hs[m * kF + ((chunk ^ (m & 7)) << 4) + lm] = (__bf16)h;
        }
      }
    }
  }
  __syncthreads();

  // ---------------- Phase B: out += scale * (h Wd^T) ----------------
  // wave owns out columns [wave*128, wave*128+128): 4 groups of 2 n-tiles
#pragma unroll 1
  for (int g = 0; g < 4; ++g) {
    const int dbase = wave * 128 + g * 32;
    v8f acc[2][2];
#pragma unroll
    for (int mt = 0; mt < 2; ++mt)
#pragma unroll
      for (int j = 0; j < 2; ++j) acc[mt][j] = vzero;

#pragma unroll 2
    for (int k0 = 0; k0 < kF; k0 += 32) {
      const int ck = ((k0 >> 4) + lh) ^ (lm & 7);
      const v16bf a0 = *(const v16bf*)(&hs[lm * kF + (ck << 4)]);
      const v16bf a1 = *(const v16bf*)(&hs[(16 + lm) * kF + (ck << 4)]);
#pragma unroll
      for (int j = 0; j < 2; ++j) {
        const float* dp = Wd + (size_t)(dbase + j * 16 + lm) * kF + lh * 16 + k0;
        const v16bf b = load_frag_f32(dp);
        acc[0][j] = wmma_bf16(a0, b, acc[0][j]);
        acc[1][j] = wmma_bf16(a1, b, acc[1][j]);
      }
    }

#pragma unroll
    for (int mt = 0; mt < 2; ++mt) {
#pragma unroll
      for (int j = 0; j < 2; ++j) {
        const int d0 = dbase + j * 16;
#pragma unroll
        for (int r = 0; r < 8; ++r) {
          const int tok = tokS[mt][r];
          if (tok >= 0)
            unsafeAtomicAdd(out + (size_t)tok * kD + d0 + lm,
                            sclS[mt][r] * acc[mt][j][r]);
        }
      }
    }
  }
}

// ---------------------------------------------------------------------------
extern "C" void kernel_launch(void* const* d_in, const int* in_sizes, int n_in,
                              void* d_out, int out_size, void* d_ws, size_t ws_size,
                              hipStream_t stream) {
  const float* x      = (const float*)d_in[0];
  const float* gate_w = (const float*)d_in[1];
  const float* ebias  = (const float*)d_in[2];
  const float* wg     = (const float*)d_in[3];
  const float* wu     = (const float*)d_in[4];
  const float* wd     = (const float*)d_in[5];
  const float* wsg    = (const float*)d_in[6];
  const float* wsu    = (const float*)d_in[7];
  const float* wsd    = (const float*)d_in[8];

  float* out = (float*)d_out;
  float* counts = out + (size_t)kT * kD;
  float* lb = counts + kE;

  char* ws = (char*)d_ws;
  int*    ecnt = (int*)(ws + OFF_ECNT);
  int*    topi = (int*)(ws + OFF_TOPI);
  float*  topw = (float*)(ws + OFF_TOPW);
  int*    list = (int*)(ws + OFF_LIST);
  float*  wl   = (float*)(ws + OFF_WL);
  __bf16* xb   = (__bf16*)(ws + OFF_XB);

  // zero output (out + counts + lb) — accumulation target
  hipMemsetAsync(d_out, 0, (size_t)out_size * sizeof(float), stream);

  // x -> bf16
  k_cvt_bf16<<<(kT * kD / 4 + 255) / 256, 256, 0, stream>>>(x, xb, kT * kD);

  // routing
  k_router<<<kT, 32, 0, stream>>>(x, gate_w, ebias, topi, topw);
  k_build<<<1, 32, 0, stream>>>(topi, topw, ecnt, list, wl, counts, lb);

  // shared expert (y == kE) + routed experts (y in [0,kE)) in ONE launch
  k_expert<<<dim3(kT / kTM, kE + 1), 256, LDS_TOTAL, stream>>>(
      xb, wg, wu, wd, wsg, wsu, wsd, list, wl, ecnt, out);
}
